// Grok1MoE_55095840473693
// MI455X (gfx1250) — matrix operand order repaired and checked by reading, not compiled, not run on hardware
//
#include <hip/hip_runtime.h>
#include <hip/hip_bf16.h>
#include <math.h>

typedef __bf16 bf16_t;
typedef __attribute__((ext_vector_type(16))) __bf16 v16bf;
typedef __attribute__((ext_vector_type(8)))  __bf16 v8bf;
typedef __attribute__((ext_vector_type(4)))  __bf16 v4bf;
typedef __attribute__((ext_vector_type(8)))  float  v8f;
typedef __attribute__((ext_vector_type(4)))  float  v4f;
typedef __attribute__((ext_vector_type(4)))  int    v4i;

#define T_TOK 4096
#define HDIM  2048
#define IDIM  4096
#define NEXP  8
#define SOFTCAP 30.0f

#define TM 128
#define TN 128
#define TK 32
#define LDSS 40   // LDS row stride in bf16 elems (80B: multiple of 16B)

// ---------- CDNA5 async global->LDS path (guarded; sync fallback) ----------
#if __has_builtin(__builtin_amdgcn_global_load_async_to_lds_b128)
#define USE_ASYNC 1
typedef __attribute__((address_space(1))) v4i gas_v4i;
typedef __attribute__((address_space(3))) v4i las_v4i;
__device__ __forceinline__ void async_copy16(const bf16_t* gsrc, bf16_t* ldst) {
    __builtin_amdgcn_global_load_async_to_lds_b128(
        (gas_v4i*)gsrc, (las_v4i*)ldst, 0, 0);
}
#if __has_builtin(__builtin_amdgcn_s_wait_asynccnt)
#define WAIT_ASYNC(n) __builtin_amdgcn_s_wait_asynccnt(n)
#else
#define WAIT_ASYNC(n) asm volatile("s_wait_asynccnt " #n)
#endif
#else
#define USE_ASYNC 0
#define WAIT_ASYNC(n)
#endif

// ---------------- utility kernels ----------------

__global__ void zero_f32_kernel(float* __restrict__ p, size_t n) {
    size_t i = (size_t)blockIdx.x * blockDim.x + threadIdx.x;
    if (i < n) p[i] = 0.0f;
}

__global__ void f32_to_bf16_kernel(const float* __restrict__ in,
                                   bf16_t* __restrict__ out, size_t n) {
    size_t i = ((size_t)blockIdx.x * blockDim.x + threadIdx.x) * 4;
    if (i + 3 < n) {
        v4f v = *(const v4f*)(in + i);
        out[i + 0] = (bf16_t)v.x;
        out[i + 1] = (bf16_t)v.y;
        out[i + 2] = (bf16_t)v.z;
        out[i + 3] = (bf16_t)v.w;
    }
}

// ---------------- router: logits -> softcap -> softmax -> top-2 ----------------

__global__ __launch_bounds__(256)
void router_kernel(const float* __restrict__ x, const float* __restrict__ gw,
                   float* __restrict__ combine) {
    const int lane = threadIdx.x & 31;
    const int wave = threadIdx.x >> 5;
    const int t = blockIdx.x * 8 + wave;
    if (t >= T_TOK) return;

    float acc[NEXP];
#pragma unroll
    for (int e = 0; e < NEXP; ++e) acc[e] = 0.0f;

    const float* xr = x + (size_t)t * HDIM;
    for (int h = lane; h < HDIM; h += 32) {
        float xv = xr[h];
        const float* gr = gw + (size_t)h * NEXP;
        v4f g0 = *(const v4f*)(gr);
        v4f g1 = *(const v4f*)(gr + 4);
        acc[0] += xv * g0.x; acc[1] += xv * g0.y;
        acc[2] += xv * g0.z; acc[3] += xv * g0.w;
        acc[4] += xv * g1.x; acc[5] += xv * g1.y;
        acc[6] += xv * g1.z; acc[7] += xv * g1.w;
    }
#pragma unroll
    for (int off = 16; off > 0; off >>= 1) {
#pragma unroll
        for (int e = 0; e < NEXP; ++e)
            acc[e] += __shfl_xor(acc[e], off, 32);
    }
    if (lane == 0) {
        float l[NEXP], m = -1e30f;
#pragma unroll
        for (int e = 0; e < NEXP; ++e) {
            l[e] = tanhf(acc[e] * (1.0f / SOFTCAP)) * SOFTCAP;
            m = fmaxf(m, l[e]);
        }
        float p[NEXP], s = 0.0f;
#pragma unroll
        for (int e = 0; e < NEXP; ++e) { p[e] = expf(l[e] - m); s += p[e]; }
        float inv = 1.0f / s;
#pragma unroll
        for (int e = 0; e < NEXP; ++e) p[e] *= inv;
        int i1 = 0;
#pragma unroll
        for (int e = 1; e < NEXP; ++e) if (p[e] > p[i1]) i1 = e;
        int i2 = (i1 == 0) ? 1 : 0;
#pragma unroll
        for (int e = 0; e < NEXP; ++e) if (e != i1 && p[e] > p[i2]) i2 = e;
        float* cr = combine + (size_t)t * NEXP;
#pragma unroll
        for (int e = 0; e < NEXP; ++e)
            cr[e] = (e == i1) ? p[i1] : ((e == i2) ? p[i2] : 0.0f);
    }
}

// ---------------- per-expert token-list compaction (deterministic order) ----------------

__global__ __launch_bounds__(256)
void build_lists_kernel(const float* __restrict__ combine,
                        int* __restrict__ lists, int* __restrict__ counts) {
    const int e = blockIdx.x;
    const int tid = threadIdx.x;
    const int lane = tid & 31;
    const int wave = tid >> 5;
    __shared__ int wtot[8];
    __shared__ int base;
    if (tid == 0) base = 0;
    __syncthreads();

    for (int start = 0; start < T_TOK; start += 256) {
        const int t = start + tid;
        const int flag = (combine[(size_t)t * NEXP + e] > 0.0f) ? 1 : 0;
        unsigned long long bal = __ballot(flag);
        unsigned b32 = (unsigned)bal;
        int lpfx = __popc(b32 & ((1u << lane) - 1u));
        if (lane == 0) wtot[wave] = __popc(b32);
        __syncthreads();
        int woff = 0, total = 0;
#pragma unroll
        for (int j = 0; j < 8; ++j) {
            int w = wtot[j];
            if (j < wave) woff += w;
            total += w;
        }
        if (flag) lists[(size_t)e * T_TOK + base + woff + lpfx] = t;
        __syncthreads();
        if (tid == 0) base += total;
        __syncthreads();
    }
    if (tid == 0) counts[e] = base;
}

// ---------------- WMMA fragment helpers ----------------

__device__ __forceinline__ v16bf frag_concat(v8bf lo, v8bf hi) {
    return __builtin_shufflevector(lo, hi, 0, 1, 2, 3, 4, 5, 6, 7,
                                           8, 9, 10, 11, 12, 13, 14, 15);
}
// A fragment: 16-bit A 16x32, contiguous K (lanes 0-15: K0-15; lanes 16-31: K16-31)
__device__ __forceinline__ v16bf load_frag_a(const bf16_t* s, int row, int ka) {
    const bf16_t* p = s + row * LDSS;
    v8bf lo = *(const v8bf*)(p + ka);
    v8bf hi = *(const v8bf*)(p + ka + 8);
    return frag_concat(lo, hi);
}
// B fragment: 16-bit B 32x16, LDS stored [N][K] => contiguous K 0-15 / 16-31
__device__ __forceinline__ v16bf load_frag_b(const bf16_t* s, int col, int kb) {
    const bf16_t* p = s + col * LDSS + kb;
    v8bf lo = *(const v8bf*)(p);
    v8bf hi = *(const v8bf*)(p + 8);
    return frag_concat(lo, hi);
}

__device__ __forceinline__ float gelu_tanh(float v) {
    float c = v + 0.044715f * v * v * v;
    return 0.5f * v * (1.0f + tanhf(0.7978845608028654f * c));
}

// ---------------- B-tile register staging: 4x4 fp32 block per thread ----------------

struct BStage { v4f r[4]; };

__device__ __forceinline__ BStage load_btile(const float* __restrict__ B, int ldb,
                                             int kk, int n0, int tid) {
    BStage s;
    const int k0 = (tid >> 5) * 4;
    const int n4 = (tid & 31) * 4;
    const float* p = B + (size_t)(kk + k0) * ldb + n0 + n4;
#pragma unroll
    for (int j = 0; j < 4; ++j) s.r[j] = *(const v4f*)(p + (size_t)j * ldb);
    return s;
}

// in-register 4x4 transpose, packed bf16 b64 stores into [N][K] LDS
__device__ __forceinline__ void store_btile(bf16_t* __restrict__ sB,
                                            const BStage& s, int tid) {
    const int k0 = (tid >> 5) * 4;
    const int n4 = (tid & 31) * 4;
#pragma unroll
    for (int c = 0; c < 4; ++c) {
        v4bf pk;
        pk.x = (bf16_t)s.r[0][c];
        pk.y = (bf16_t)s.r[1][c];
        pk.z = (bf16_t)s.r[2][c];
        pk.w = (bf16_t)s.r[3][c];
        *(v4bf*)(&sB[(n4 + c) * LDSS + k0]) = pk;
    }
}

// ---------------- GEMM 1&3 fused: g = gelu(x@W1) * (x@W3), gathered rows ----------------

__global__ __launch_bounds__(256)
void gemm13_kernel(const bf16_t* __restrict__ xbf,
                   const float* __restrict__ w1e,
                   const float* __restrict__ w3e,
                   const int* __restrict__ list_e,
                   const int* __restrict__ cnt_ptr,
                   bf16_t* __restrict__ g) {
    __shared__ __align__(16) bf16_t sA[2][TM * LDSS];
    __shared__ __align__(16) bf16_t sB1[2][TN * LDSS];
    __shared__ __align__(16) bf16_t sB3[2][TN * LDSS];
    __shared__ int ridx[TM];

    const int cnt = *cnt_ptr;
    const int m0 = blockIdx.y * TM;
    if (m0 >= cnt) return;
    const int n0 = blockIdx.x * TN;
    const int tid = threadIdx.x;

    if (tid < TM) {
        int r = m0 + tid;
        ridx[tid] = list_e[(r < cnt) ? r : (cnt - 1)];
    }
    __syncthreads();

    const int lane = tid & 31;
    const int wave = tid >> 5;
    const int wm = (wave & 1) * 64;
    const int wn = (wave >> 1) * 32;
    const bool lo = lane < 16;
    const int lmod = lane & 15;
    const int ka = lo ? 0 : 16;
    const int kb = lo ? 0 : 16;

    v8f acc1[4][2], acc3[4][2];
    const v8f vz = {0, 0, 0, 0, 0, 0, 0, 0};
#pragma unroll
    for (int mi = 0; mi < 4; ++mi)
#pragma unroll
        for (int ni = 0; ni < 2; ++ni) { acc1[mi][ni] = vz; acc3[mi][ni] = vz; }

    // per-thread A-fill coordinates (16B chunks)
    const int ar = tid >> 1;
    const int ahalf = (tid & 1) * 16;
    const bf16_t* arow = xbf + (size_t)ridx[ar] * HDIM + ahalf;

    const int NK = HDIM / TK;

    // ---- prologue: tile 0 in flight ----
    BStage b1s = load_btile(w1e, IDIM, 0, n0, tid);
    BStage b3s = load_btile(w3e, IDIM, 0, n0, tid);
#if USE_ASYNC
    async_copy16(arow, &sA[0][ar * LDSS + ahalf]);
    async_copy16(arow + 8, &sA[0][ar * LDSS + ahalf + 8]);
#else
    v8bf a0 = ((const v8bf*)arow)[0];
    v8bf a1 = ((const v8bf*)arow)[1];
#endif

    for (int i = 0; i < NK; ++i) {
        const int cur = i & 1;
        const int nxt = cur ^ 1;
        const int kknxt = (i + 1) * TK;

        store_btile(sB1[cur], b1s, tid);
        store_btile(sB3[cur], b3s, tid);
#if !USE_ASYNC
        *(v8bf*)(&sA[cur][ar * LDSS + ahalf]) = a0;
        *(v8bf*)(&sA[cur][ar * LDSS + ahalf + 8]) = a1;
#endif
        if (i + 1 < NK) {
#if USE_ASYNC
            async_copy16(arow + kknxt, &sA[nxt][ar * LDSS + ahalf]);
            async_copy16(arow + kknxt + 8, &sA[nxt][ar * LDSS + ahalf + 8]);
#endif
            b1s = load_btile(w1e, IDIM, kknxt, n0, tid);
            b3s = load_btile(w3e, IDIM, kknxt, n0, tid);
#if !USE_ASYNC
            a0 = ((const v8bf*)(arow + kknxt))[0];
            a1 = ((const v8bf*)(arow + kknxt))[1];
#endif
            WAIT_ASYNC(2);   // drain tile i's async A loads (in-order per wave)
        } else {
            WAIT_ASYNC(0);
        }
        __syncthreads();

        v16bf af[4], b1f[2], b3f[2];
#pragma unroll
        for (int mi = 0; mi < 4; ++mi)
            af[mi] = load_frag_a(sA[cur], wm + mi * 16 + lmod, ka);
#pragma unroll
        for (int ni = 0; ni < 2; ++ni) {
            b1f[ni] = load_frag_b(sB1[cur], wn + ni * 16 + lmod, kb);
            b3f[ni] = load_frag_b(sB3[cur], wn + ni * 16 + lmod, kb);
        }
#pragma unroll
        for (int mi = 0; mi < 4; ++mi)
#pragma unroll
            for (int ni = 0; ni < 2; ++ni) {
                acc1[mi][ni] = __builtin_amdgcn_wmma_f32_16x16x32_bf16(
                    false, af[mi], false, b1f[ni], (short)0, acc1[mi][ni], false, false);
                acc3[mi][ni] = __builtin_amdgcn_wmma_f32_16x16x32_bf16(
                    false, af[mi], false, b3f[ni], (short)0, acc3[mi][ni], false, false);
            }
        __syncthreads();
    }

    // epilogue: h = gelu(c1) * c3 -> bf16 g[compact_row][n]
#pragma unroll
    for (int mi = 0; mi < 4; ++mi)
#pragma unroll
        for (int ni = 0; ni < 2; ++ni)
#pragma unroll
            for (int v = 0; v < 8; ++v) {
                int row = wm + mi * 16 + v + (lo ? 0 : 8);
                int rg = m0 + row;
                if (rg < cnt) {
                    int n = n0 + wn + ni * 16 + lmod;
                    float h = gelu_tanh(acc1[mi][ni][v]) * acc3[mi][ni][v];
                    g[(size_t)rg * IDIM + n] = (bf16_t)h;
                }
            }
}

// ---------------- GEMM 2: out[t] += combine[t,e] * (g @ W2) ----------------

__global__ __launch_bounds__(256)
void gemm2_kernel(const bf16_t* __restrict__ g,
                  const float* __restrict__ w2e,
                  const int* __restrict__ list_e,
                  const int* __restrict__ cnt_ptr,
                  const float* __restrict__ combine,
                  int e,
                  float* __restrict__ out) {
    __shared__ __align__(16) bf16_t sA[2][TM * LDSS];
    __shared__ __align__(16) bf16_t sB[2][TN * LDSS];

    const int cnt = *cnt_ptr;
    const int m0 = blockIdx.y * TM;
    if (m0 >= cnt) return;
    const int n0 = blockIdx.x * TN;
    const int tid = threadIdx.x;

    const int lane = tid & 31;
    const int wave = tid >> 5;
    const int wm = (wave & 1) * 64;
    const int wn = (wave >> 1) * 32;
    const bool lo = lane < 16;
    const int lmod = lane & 15;
    const int ka = lo ? 0 : 16;
    const int kb = lo ? 0 : 16;

    v8f acc[4][2];
    const v8f vz = {0, 0, 0, 0, 0, 0, 0, 0};
#pragma unroll
    for (int mi = 0; mi < 4; ++mi)
#pragma unroll
        for (int ni = 0; ni < 2; ++ni) acc[mi][ni] = vz;

    const int ar = tid >> 1;
    const int ahalf = (tid & 1) * 16;
    // compact rows: rows >= cnt hold stale data but never affect valid rows
    const bf16_t* arow = g + (size_t)(m0 + ar) * IDIM + ahalf;

    const int NK = IDIM / TK;

    BStage bs = load_btile(w2e, HDIM, 0, n0, tid);
#if USE_ASYNC
    async_copy16(arow, &sA[0][ar * LDSS + ahalf]);
    async_copy16(arow + 8, &sA[0][ar * LDSS + ahalf + 8]);
#else
    v8bf a0 = ((const v8bf*)arow)[0];
    v8bf a1 = ((const v8bf*)arow)[1];
#endif

    for (int i = 0; i < NK; ++i) {
        const int cur = i & 1;
        const int nxt = cur ^ 1;
        const int kknxt = (i + 1) * TK;

        store_btile(sB[cur], bs, tid);
#if !USE_ASYNC
        *(v8bf*)(&sA[cur][ar * LDSS + ahalf]) = a0;
        *(v8bf*)(&sA[cur][ar * LDSS + ahalf + 8]) = a1;
#endif
        if (i + 1 < NK) {
#if USE_ASYNC
            async_copy16(arow + kknxt, &sA[nxt][ar * LDSS + ahalf]);
            async_copy16(arow + kknxt + 8, &sA[nxt][ar * LDSS + ahalf + 8]);
#endif
            bs = load_btile(w2e, HDIM, kknxt, n0, tid);
#if !USE_ASYNC
            a0 = ((const v8bf*)(arow + kknxt))[0];
            a1 = ((const v8bf*)(arow + kknxt))[1];
#endif
            WAIT_ASYNC(2);
        } else {
            WAIT_ASYNC(0);
        }
        __syncthreads();

        v16bf af[4], bf[2];
#pragma unroll
        for (int mi = 0; mi < 4; ++mi)
            af[mi] = load_frag_a(sA[cur], wm + mi * 16 + lmod, ka);
#pragma unroll
        for (int ni = 0; ni < 2; ++ni)
            bf[ni] = load_frag_b(sB[cur], wn + ni * 16 + lmod, kb);
#pragma unroll
        for (int mi = 0; mi < 4; ++mi)
#pragma unroll
            for (int ni = 0; ni < 2; ++ni)
                acc[mi][ni] = __builtin_amdgcn_wmma_f32_16x16x32_bf16(
                    false, af[mi], false, bf[ni], (short)0, acc[mi][ni], false, false);
        __syncthreads();
    }

    // epilogue: scatter-add scaled rows (expert kernels are stream-serialized)
#pragma unroll
    for (int mi = 0; mi < 4; ++mi)
#pragma unroll
        for (int ni = 0; ni < 2; ++ni)
#pragma unroll
            for (int v = 0; v < 8; ++v) {
                int row = wm + mi * 16 + v + (lo ? 0 : 8);
                int rg = m0 + row;
                if (rg < cnt) {
                    int t = list_e[rg];
                    float sc = combine[(size_t)t * NEXP + e];
                    int n = n0 + wn + ni * 16 + lmod;
                    out[(size_t)t * HDIM + n] += sc * acc[mi][ni][v];
                }
            }
}

// ---------------- host ----------------

extern "C" void kernel_launch(void* const* d_in, const int* in_sizes, int n_in,
                              void* d_out, int out_size, void* d_ws, size_t ws_size,
                              hipStream_t stream) {
    const float* x  = (const float*)d_in[0];
    const float* gw = (const float*)d_in[1];
    const float* w1 = (const float*)d_in[2];
    const float* w3 = (const float*)d_in[3];
    const float* w2 = (const float*)d_in[4];
    float* out = (float*)d_out;

    char* ws = (char*)d_ws;
    float* combine = (float*)ws;            ws += (size_t)T_TOK * NEXP * sizeof(float);
    int*   counts  = (int*)ws;              ws += 256;
    int*   lists   = (int*)ws;              ws += (size_t)NEXP * T_TOK * sizeof(int);
    bf16_t* xbf    = (bf16_t*)ws;           ws += (size_t)T_TOK * HDIM * sizeof(bf16_t);
    bf16_t* g      = (bf16_t*)ws;           // T_TOK * IDIM bf16 (reused per expert)

    const size_t nout = (size_t)T_TOK * HDIM;
    zero_f32_kernel<<<(unsigned)((nout + 255) / 256), 256, 0, stream>>>(out, nout);
    router_kernel<<<T_TOK / 8, 256, 0, stream>>>(x, gw, combine);
    build_lists_kernel<<<NEXP, 256, 0, stream>>>(combine, lists, counts);
    f32_to_bf16_kernel<<<(unsigned)(nout / 1024), 256, 0, stream>>>(x, xbf, nout);

    for (int e = 0; e < NEXP; ++e) {
        const float* w1e = w1 + (size_t)e * HDIM * IDIM;
        const float* w3e = w3 + (size_t)e * HDIM * IDIM;
        const float* w2e = w2 + (size_t)e * IDIM * HDIM;
        const int* le = lists + (size_t)e * T_TOK;
        gemm13_kernel<<<dim3(IDIM / TN, T_TOK / TM), 256, 0, stream>>>(
            xbf, w1e, w3e, le, counts + e, g);
        gemm2_kernel<<<dim3(HDIM / TN, T_TOK / TM), 256, 0, stream>>>(
            g, w2e, le, counts + e, combine, e, out);
    }
}
